// MLP_17308718202952
// MI455X (gfx1250) — compile-verified, matrix operand order checked by reading, not run on hardware
//
#include <hip/hip_runtime.h>

// ---------------------------------------------------------------- types
typedef __attribute__((ext_vector_type(16))) __bf16 v16bf;
typedef __attribute__((ext_vector_type(8)))  float  v8f;
typedef __attribute__((ext_vector_type(4)))  unsigned int v4u;
typedef __attribute__((ext_vector_type(8)))  unsigned int v8u;

#define N_NODES 40000
#define D_IN    512
#define D_OUT   256
#define KNN     33            // k+1
#define NPAD    40192         // 157 * 256
#define NBLK    157           // ceil(40000/256) column blocks
#define E_EDGES (N_NODES * KNN)

// ---------------------------------------------------------------- helpers
__device__ __forceinline__ unsigned short f32_to_bf16(float x) {
  unsigned int u = __float_as_uint(x);
  unsigned int r = u + 0x7FFFu + ((u >> 16) & 1u);   // round-to-nearest-even
  return (unsigned short)(r >> 16);
}
__device__ __forceinline__ float bf16_to_f32(unsigned short h) {
  return __uint_as_float(((unsigned int)h) << 16);
}

__device__ __forceinline__ v8f wmma_bf16(v16bf a, v16bf b, v8f c) {
  return __builtin_amdgcn_wmma_f32_16x16x32_bf16(false, a, false, b, (short)0, c,
                                                 false, false);
}

// Async global->LDS copy, 128 bits per lane (GVS mode: saddr base + 32b voffset).
__device__ __forceinline__ void async_b128(unsigned ldsByte,
                                           const void* base,
                                           unsigned gByteOff) {
  asm volatile("global_load_async_to_lds_b128 %0, %1, %2"
               :
               : "v"(ldsByte), "v"(gByteOff),
                 "s"((unsigned long long)(size_t)base)
               : "memory");
}
__device__ __forceinline__ void wait_async8() {
  asm volatile("s_wait_asynccnt 0x8" ::: "memory");
}
__device__ __forceinline__ void wait_async0() {
  asm volatile("s_wait_asynccnt 0x0" ::: "memory");
}

// TDM: DMA a 2D tile (256 bf16 wide x `rows` tall) from a row-major NPAD x 256
// bf16 tensor into LDS. D# group0/group1 built per CDNA5 ISA 8.3/8.4.
// Tracked by TENSORcnt.
__device__ __forceinline__ void tdm_load_2d(unsigned ldsByte, const void* gbase,
                                            unsigned rows) {
  unsigned long long ga = (unsigned long long)(size_t)gbase;
  v4u g0;
  g0.x = 1u;                                   // count=1 (valid), user mode
  g0.y = ldsByte;                              // lds_addr
  g0.z = (unsigned)ga;                         // global_addr[31:0]
  g0.w = (unsigned)(ga >> 32) | 0x80000000u;   // global_addr[56:32] | type=2
  v8u g1;
  g1.s0 = 0x00010000u;                         // data_size=1 (2 bytes)
  g1.s1 = (256u & 0xFFFFu) << 16;              // tensor_dim0 = 256 [47:32]->[79:48]
  g1.s2 = ((unsigned)NPAD & 0xFFFFu) << 16;    // tensor_dim1 lo16
  g1.s3 = ((unsigned)NPAD >> 16) | (256u << 16); // tensor_dim1 hi | tile_dim0=256
  g1.s4 = rows;                                // tile_dim1 = rows, tile_dim2 = 0
  g1.s5 = 256u;                                // tensor_dim0_stride = 256 elems
  g1.s6 = 0u;
  g1.s7 = 0u;
  asm volatile("tensor_load_to_lds %0, %1" :: "s"(g0), "s"(g1) : "memory");
}

// Load a 16x32 bf16 fragment (A or B mirror layout) from LDS.
// Per ISA layout: lane = h*16 + r; VGPR v holds K-pair kp = (v&3) + h*4 + (v>>2)*8.
__device__ __forceinline__ v16bf load_frag(const unsigned int* base, int row0,
                                           int strideDw, int kDw) {
  int lane = threadIdx.x & 31;
  int h = lane >> 4;
  int r = lane & 15;
  union { unsigned int u[8]; v16bf v; } f;
#pragma unroll
  for (int vv = 0; vv < 8; ++vv) {
    int kp = (vv & 3) + h * 4 + (vv >> 2) * 8;
    f.u[vv] = base[(row0 + r) * strideDw + kDw + kp];
  }
  return f.v;
}

// ---------------------------------------------------------------- 1) f32 -> bf16 hi/lo
__global__ void decompose_kernel(const float* __restrict__ src,
                                 unsigned short* __restrict__ hi,
                                 unsigned short* __restrict__ lo, int n) {
  for (int i = blockIdx.x * blockDim.x + threadIdx.x; i < n;
       i += gridDim.x * blockDim.x) {
    float x = src[i];
    unsigned short h = f32_to_bf16(x);
    hi[i] = h;
    lo[i] = f32_to_bf16(x - bf16_to_f32(h));
  }
}

// ---------------------------------------------------------------- 2/3) bf16x2 GEMM
template <int KDIM, bool RELU, bool DECOMP_OUT>
__global__ __launch_bounds__(256) void gemm_bf16x2_kernel(
    const unsigned short* __restrict__ Ahi, const unsigned short* __restrict__ Alo,
    const unsigned short* __restrict__ Bhi, const unsigned short* __restrict__ Blo,
    const float* __restrict__ bias,
    unsigned short* __restrict__ OutHi, unsigned short* __restrict__ OutLo,
    float* __restrict__ OutF, int Nn) {
  __shared__ __align__(16) unsigned int As[2][32][16];  // [hi/lo][m][kpair]
  __shared__ __align__(16) unsigned int Bs[2][64][16];  // [hi/lo][n][kpair]

  const unsigned int* Ahi_u = (const unsigned int*)Ahi;
  const unsigned int* Alo_u = (const unsigned int*)Alo;
  const unsigned int* Bhi_u = (const unsigned int*)Bhi;
  const unsigned int* Blo_u = (const unsigned int*)Blo;

  const int m0 = blockIdx.x * 32;
  const int n0 = blockIdx.y * 64;
  const int wave = threadIdx.x >> 5;
  const int mi = wave >> 2;  // 0..1
  const int ni = wave & 3;   // 0..3

  v8f acc = {0.f, 0.f, 0.f, 0.f, 0.f, 0.f, 0.f, 0.f};

  for (int k0 = 0; k0 < KDIM; k0 += 32) {
    __syncthreads();
    const int kdw = k0 >> 1;
    for (int idx = threadIdx.x; idx < 1024; idx += 256) {
      int sel = idx >> 9, rem = idx & 511;
      int row = rem >> 4, kp = rem & 15;
      const unsigned int* s = sel ? Alo_u : Ahi_u;
      (&As[0][0][0])[idx] = s[(size_t)(m0 + row) * (KDIM / 2) + kdw + kp];
    }
    for (int idx = threadIdx.x; idx < 2048; idx += 256) {
      int sel = idx >> 10, rem = idx & 1023;
      int row = rem >> 4, kp = rem & 15;
      const unsigned int* s = sel ? Blo_u : Bhi_u;
      (&Bs[0][0][0])[idx] = s[(size_t)(n0 + row) * (KDIM / 2) + kdw + kp];
    }
    __syncthreads();
    v16bf ah = load_frag(&As[0][0][0], mi * 16, 16, 0);
    v16bf al = load_frag(&As[1][0][0], mi * 16, 16, 0);
    v16bf bh = load_frag(&Bs[0][0][0], ni * 16, 16, 0);
    v16bf bl = load_frag(&Bs[1][0][0], ni * 16, 16, 0);
    acc = wmma_bf16(ah, bh, acc);
    acc = wmma_bf16(ah, bl, acc);
    acc = wmma_bf16(al, bh, acc);
  }

  const int lane = threadIdx.x & 31;
  const int h = lane >> 4;
  const int nl = lane & 15;
  const int gm0 = m0 + mi * 16 + h * 8;
  const int gn = n0 + ni * 16 + nl;
  const float bv = bias[gn];
#pragma unroll
  for (int v = 0; v < 8; ++v) {
    float val = acc[v] + bv;
    if (RELU) val = fmaxf(val, 0.f);
    size_t off = (size_t)(gm0 + v) * Nn + gn;
    if constexpr (DECOMP_OUT) {
      unsigned short hh = f32_to_bf16(val);
      OutHi[off] = hh;
      OutLo[off] = f32_to_bf16(val - bf16_to_f32(hh));
    } else {
      OutF[off] = val;
    }
  }
}

// ---------------------------------------------------------------- 4) row normalize
__global__ __launch_bounds__(256) void normalize_kernel(
    const float* __restrict__ emb, unsigned short* __restrict__ xh,
    unsigned short* __restrict__ xl) {
  __shared__ float red[256];
  const int row = blockIdx.x;
  const int t = threadIdx.x;
  float v = emb[(size_t)row * D_OUT + t];
  red[t] = v * v;
  __syncthreads();
  for (int s = 128; s > 0; s >>= 1) {
    if (t < s) red[t] += red[t + s];
    __syncthreads();
  }
  float nrm = fmaxf(sqrtf(red[0]), 1e-12f);
  float x = v / nrm;
  unsigned short h = f32_to_bf16(x);
  xh[(size_t)row * D_OUT + t] = h;
  xl[(size_t)row * D_OUT + t] = f32_to_bf16(x - bf16_to_f32(h));
}

// ---------------------------------------------------------------- 5) similarity + top-k
// WG = 256 threads (8 waves) handles 64 rows. 157 column blocks of 256.
// A tile staged once via TDM (TENSORcnt); B k-chunks via double-buffered async
// global->LDS (ASYNCcnt) so the fetch of chunk k+1 overlaps the WMMAs of chunk k.
union __align__(16) SimsBs {
  unsigned int bs[2][2][256][16];  // [buf][hi/lo][col][kpair] : 2 x 32 KB
  float sims[64][256];             // similarity tile (aliased, disjoint in time)
};

__global__ __launch_bounds__(256) void simtopk_kernel(
    const unsigned short* __restrict__ Xhi, const unsigned short* __restrict__ Xlo,
    float* __restrict__ topv, int* __restrict__ topi) {
  __shared__ __align__(16) unsigned int As[2][64][128];  // 64 rows, K=256 (hi/lo)
  __shared__ SimsBs SB;
  __shared__ float lv[256 * KNN];
  __shared__ int li[256 * KNN];

  const int m0 = blockIdx.x * 64;
  const int t = threadIdx.x;
  const int wave = t >> 5;
  const int lane = t & 31;
  const int h = lane >> 4;
  const int nl = lane & 15;
  const int mpair = wave >> 2;  // 0..1 : m-tiles {2*mpair, 2*mpair+1}
  const int nq = wave & 3;      // n-tiles {nq, nq+4, nq+8, nq+12}

  const unsigned asBase = (unsigned)(size_t)&As[0][0][0];
  const unsigned bsBase0 = (unsigned)(size_t)&SB.bs[0][0][0][0];
  const unsigned bsBase1 = (unsigned)(size_t)&SB.bs[1][0][0][0];

  // ---- stage A tile (64 x 256 bf16, hi & lo planes) via TDM, wave 0 only
  if (wave == 0) {
    tdm_load_2d(asBase, Xhi + (size_t)m0 * D_OUT, 64u);
    tdm_load_2d(asBase + 64u * 512u, Xlo + (size_t)m0 * D_OUT, 64u);
  }

  // init private top lists (4 scan threads per row)
  float minVal = -2.0f;
  int minPos = 0;
  {
    const int base = t * KNN;
    for (int q = 0; q < KNN; ++q) { lv[base + q] = -2.0f; li[base + q] = 0; }
  }

  if (wave == 0) __builtin_amdgcn_s_wait_tensorcnt(0);
  __syncthreads();  // A tile resident for all waves

  for (int cb = 0; cb < NBLK; ++cb) {
    v8f acc[2][4];
#pragma unroll
    for (int mt = 0; mt < 2; ++mt)
#pragma unroll
      for (int j = 0; j < 4; ++j)
        acc[mt][j] = (v8f){0.f, 0.f, 0.f, 0.f, 0.f, 0.f, 0.f, 0.f};

    // issue B stage for kc=0 into buffer 0 (8 insts/wave)
#pragma unroll
    for (int j = 0; j < 8; ++j) {
      int c = t + 256 * j;
      int row = (c & 1023) >> 2;
      int kp4 = (c & 3) * 4;
      unsigned gOff = (unsigned)(((cb * 256 + row) * 128 + 0 * 16 + kp4) * 4);
      async_b128(bsBase0 + (unsigned)c * 16u, (j >> 2) ? Xlo : Xhi, gOff);
    }

    for (int kc = 0; kc < 8; ++kc) {
      const int buf = kc & 1;
      if (kc + 1 < 8) {  // prefetch next chunk into other buffer
        unsigned dstBase = (buf ^ 1) ? bsBase1 : bsBase0;
#pragma unroll
        for (int j = 0; j < 8; ++j) {
          int c = t + 256 * j;
          int row = (c & 1023) >> 2;
          int kp4 = (c & 3) * 4;
          unsigned gOff =
              (unsigned)(((cb * 256 + row) * 128 + (kc + 1) * 16 + kp4) * 4);
          async_b128(dstBase + (unsigned)c * 16u, (j >> 2) ? Xlo : Xhi, gOff);
        }
        wait_async8();  // chunk kc landed; only kc+1 outstanding
      } else {
        wait_async0();
      }
      __syncthreads();  // all waves' chunk-kc data visible

      v16bf ah[2], al[2];
#pragma unroll
      for (int mt = 0; mt < 2; ++mt) {
        ah[mt] = load_frag(&As[0][0][0], (mpair * 2 + mt) * 16, 128, kc * 16);
        al[mt] = load_frag(&As[1][0][0], (mpair * 2 + mt) * 16, 128, kc * 16);
      }
#pragma unroll
      for (int j = 0; j < 4; ++j) {
        int nt = nq + j * 4;
        v16bf bh = load_frag(&SB.bs[buf][0][0][0], nt * 16, 16, 0);
        v16bf bl = load_frag(&SB.bs[buf][1][0][0], nt * 16, 16, 0);
#pragma unroll
        for (int mt = 0; mt < 2; ++mt) {
          acc[mt][j] = wmma_bf16(ah[mt], bh, acc[mt][j]);
          acc[mt][j] = wmma_bf16(ah[mt], bl, acc[mt][j]);
          acc[mt][j] = wmma_bf16(al[mt], bh, acc[mt][j]);
        }
      }
      __syncthreads();  // all reads of buf done before it is overwritten
    }

    // write sims (aliases Bs; all async retired and all waves past barrier)
#pragma unroll
    for (int mt = 0; mt < 2; ++mt) {
      int mrow = (mpair * 2 + mt) * 16;
#pragma unroll
      for (int j = 0; j < 4; ++j) {
        int nt = nq + j * 4;
#pragma unroll
        for (int v = 0; v < 8; ++v)
          SB.sims[mrow + v + 8 * h][nt * 16 + nl] = acc[mt][j][v];
      }
    }
    __syncthreads();

    // scan: 4 threads per row maintain private top-33 lists in LDS
    {
      const int row = t >> 2;
      const int l4 = t & 3;
      const int base = t * KNN;
      for (int j = 0; j < 64; ++j) {
        int coll = l4 + j * 4;
        int gcol = cb * 256 + coll;
        if (gcol < N_NODES) {
          float c = SB.sims[row][coll];
          if (c > minVal) {
            lv[base + minPos] = c;
            li[base + minPos] = gcol;
            float mv = lv[base]; int mp = 0;
            for (int q = 1; q < KNN; ++q) {
              float xq = lv[base + q];
              if (xq < mv) { mv = xq; mp = q; }
            }
            minVal = mv; minPos = mp;
          }
        }
      }
    }
    __syncthreads();  // scans done before next cb stages into aliased region
  }

  // merge: thread r < 64 extracts top-33 (descending) from its row's 4*33 entries
  if (t < 64) {
    const int base = t * 4 * KNN;  // 132 candidates
    for (int s = 0; s < KNN; ++s) {
      float best = -3.0f; int bpos = 0;
      for (int q = 0; q < 4 * KNN; ++q) {
        float xq = lv[base + q];
        if (xq > best) { best = xq; bpos = q; }
      }
      topv[(size_t)(m0 + t) * KNN + s] = best;
      topi[(size_t)(m0 + t) * KNN + s] = li[base + bpos];
      lv[base + bpos] = -3.0f;
    }
  }
}

// ---------------------------------------------------------------- 6) norm = row+col sums
__global__ void norm_accum_kernel(const float* __restrict__ topv,
                                  const int* __restrict__ topi,
                                  float* __restrict__ nrm) {
  int e = blockIdx.x * blockDim.x + threadIdx.x;
  if (e < E_EDGES) {
    float v = topv[e];
    atomicAdd(&nrm[e / KNN], v);    // norm_row
    atomicAdd(&nrm[topi[e]], v);    // norm_col (segment sum)
  }
}

// ---------------------------------------------------------------- 7) finalize outputs
__global__ void finalize_kernel(const float* __restrict__ topv,
                                const int* __restrict__ topi,
                                const float* __restrict__ nrm,
                                float* __restrict__ out) {
  int e = blockIdx.x * blockDim.x + threadIdx.x;
  if (e < E_EDGES) {
    int r = e / KNN;
    int c = topi[e];
    float w = topv[e] * rsqrtf(nrm[r]) * rsqrtf(nrm[c]);
    const size_t E = E_EDGES;
    out[0 * E + e] = (float)r;
    out[1 * E + e] = (float)c;
    out[2 * E + e] = (float)c;
    out[3 * E + e] = (float)r;
    out[4 * E + e] = w;
    out[5 * E + e] = w;
  }
}

// ---------------------------------------------------------------- launcher
extern "C" void kernel_launch(void* const* d_in, const int* in_sizes, int n_in,
                              void* d_out, int out_size, void* d_ws, size_t ws_size,
                              hipStream_t stream) {
  (void)in_sizes; (void)n_in; (void)out_size; (void)ws_size;
  const float* x  = (const float*)d_in[0];
  const float* W1 = (const float*)d_in[1];
  const float* b1 = (const float*)d_in[2];
  const float* W2 = (const float*)d_in[3];
  const float* b2 = (const float*)d_in[4];
  float* out = (float*)d_out;

  char* ws = (char*)d_ws;
  size_t off = 0;
  auto alloc = [&](size_t bytes) -> char* {
    char* p = ws + off;
    off = (off + bytes + 255) & ~(size_t)255;
    return p;
  };

  unsigned short* xh  = (unsigned short*)alloc((size_t)N_NODES * D_IN * 2);
  unsigned short* xl  = (unsigned short*)alloc((size_t)N_NODES * D_IN * 2);
  unsigned short* w1h = (unsigned short*)alloc((size_t)D_IN * D_IN * 2);
  unsigned short* w1l = (unsigned short*)alloc((size_t)D_IN * D_IN * 2);
  unsigned short* w2h = (unsigned short*)alloc((size_t)D_OUT * D_IN * 2);
  unsigned short* w2l = (unsigned short*)alloc((size_t)D_OUT * D_IN * 2);
  unsigned short* hh  = (unsigned short*)alloc((size_t)N_NODES * D_IN * 2);
  unsigned short* hl  = (unsigned short*)alloc((size_t)N_NODES * D_IN * 2);
  float*          emb = (float*)alloc((size_t)N_NODES * D_OUT * 4);
  unsigned short* xnh = (unsigned short*)alloc((size_t)NPAD * D_OUT * 2);
  unsigned short* xnl = (unsigned short*)alloc((size_t)NPAD * D_OUT * 2);
  float*          tv  = (float*)alloc((size_t)E_EDGES * 4);
  int*            ti  = (int*)alloc((size_t)E_EDGES * 4);
  float*          nrm = (float*)alloc((size_t)N_NODES * 4);

  decompose_kernel<<<2048, 256, 0, stream>>>(x, xh, xl, N_NODES * D_IN);
  decompose_kernel<<<256, 256, 0, stream>>>(W1, w1h, w1l, D_IN * D_IN);
  decompose_kernel<<<128, 256, 0, stream>>>(W2, w2h, w2l, D_OUT * D_IN);

  gemm_bf16x2_kernel<D_IN, true, true>
      <<<dim3(N_NODES / 32, D_IN / 64), 256, 0, stream>>>(
          xh, xl, w1h, w1l, b1, hh, hl, nullptr, D_IN);

  gemm_bf16x2_kernel<D_IN, false, false>
      <<<dim3(N_NODES / 32, D_OUT / 64), 256, 0, stream>>>(
          hh, hl, w2h, w2l, b2, nullptr, nullptr, emb, D_OUT);

  normalize_kernel<<<N_NODES, 256, 0, stream>>>(emb, xnh, xnl);

  simtopk_kernel<<<N_NODES / 64, 256, 0, stream>>>(xnh, xnl, tv, ti);

  hipMemsetAsync(nrm, 0, (size_t)N_NODES * 4, stream);
  norm_accum_kernel<<<(E_EDGES + 255) / 256, 256, 0, stream>>>(tv, ti, nrm);

  finalize_kernel<<<(E_EDGES + 255) / 256, 256, 0, stream>>>(tv, ti, nrm, out);
}
